// Generator_13151189860628
// MI455X (gfx1250) — matrix operand order repaired and checked by reading, not compiled, not run on hardware
//
#include <hip/hip_runtime.h>
#include <hip/hip_bf16.h>

// ---------------- problem constants (from reference) ----------------
#define Hh      256
#define H3      768
#define LATENT  128
#define Bb      512
#define EDGE_W  12
#define MAXN    50
#define END_NODE 39
#define RING    13          // 12-window + 1 so writer never clobbers a reader
#define INV_TEMP 1000.0f    // 1/1e-3
#define WPAD    56          // LDS row stride (48 cols + 8 pad) -> 112B, 16B aligned

typedef __attribute__((ext_vector_type(16))) __bf16 v16bf;
typedef __attribute__((ext_vector_type(8)))  float  v8f;

struct BF16Frag { uint4 lo; uint4 hi; };   // 32 bytes == v16bf

// ---------------- helpers ----------------
__device__ __forceinline__ unsigned short f2bf_bits(float f) {
    unsigned int u = __builtin_bit_cast(unsigned int, f);
    unsigned int lsb = (u >> 16) & 1u;
    u += 0x7FFFu + lsb;                    // round-to-nearest-even
    return (unsigned short)(u >> 16);
}

// A-fragment, 16-bit 16x32 layout: lane holds row M=lane&15; K groups
// {half*8 .. half*8+7} and {16+half*8 .. 16+half*8+7}. p points at first group.
__device__ __forceinline__ v16bf load_frag_a(const unsigned short* p) {
    BF16Frag r;
    r.lo = *(const uint4*)(p);
    r.hi = *(const uint4*)(p + 16);
    return __builtin_bit_cast(v16bf, r);
}
// B-fragment, 32x16 bf16: lane holds N=lane&15; K groups {half*8..+7} and
// {16+half*8..+7} (row stride ld). p points at (K=half*8, N).
__device__ __forceinline__ v16bf load_frag_b(const unsigned short* p, int ld) {
    struct { unsigned short t[16]; } r;
    for (int i = 0; i < 8; ++i) {
        r.t[i]     = p[i * ld];
        r.t[8 + i] = p[(16 + i) * ld];
    }
    return __builtin_bit_cast(v16bf, r);
}
__device__ __forceinline__ v8f wmma_bf16(v16bf a, v16bf b, v8f c) {
    return __builtin_amdgcn_wmma_f32_16x16x32_bf16(
        /*neg_a=*/false, a, /*neg_b=*/false, b,
        /*c_mod=*/(short)0, c, /*reuse_a=*/false, /*reuse_b=*/false);
}

__device__ __forceinline__ float u01(unsigned long long seed, unsigned long long idx) {
    unsigned long long x = seed * 0x9E3779B97F4A7C15ull + idx + 0x2545F4914F6CDD1Dull;
    x ^= x >> 30; x *= 0xBF58476D1CE4E5B9ull;
    x ^= x >> 27; x *= 0x94D049BB133111EBull;
    x ^= x >> 31;
    return ((float)(x >> 40) + 0.5f) * (1.0f / 16777216.0f);
}
__device__ __forceinline__ float gumbel(unsigned long long seed, unsigned long long idx) {
    return -__logf(-__logf(u01(seed, idx)));
}
__device__ __forceinline__ float sigmoidf(float x) { return 1.0f / (1.0f + __expf(-x)); }

// ---------------- conversion / init kernels ----------------
// Wt[k*OUT + o] = bf16(W[o*IN + k])  (K-major transpose for B-fragments)
__global__ void convT_kernel(const float* __restrict__ W, unsigned short* __restrict__ Wt,
                             int OUT, int IN) {
    int i = blockIdx.x * blockDim.x + threadIdx.x;
    if (i < OUT * IN) {
        int o = i / IN, k = i - o * IN;
        Wt[k * OUT + o] = f2bf_bits(W[i]);
    }
}
__global__ void conv_kernel(const float* __restrict__ a, unsigned short* __restrict__ o, int n) {
    int i = blockIdx.x * blockDim.x + threadIdx.x;
    if (i < n) o[i] = f2bf_bits(a[i]);
}
__global__ void init_kernel(float* num_nodes, unsigned short* x0,
                            float* ehf, unsigned short* ehb) {
    int i = blockIdx.x * blockDim.x + threadIdx.x;
    if (i < Bb) num_nodes[i] = -1.0f;
    if (i < Bb * Hh) { x0[i] = 0; ehf[i] = 0.0f; ehb[i] = 0; }
}
__global__ void finalize_kernel(float* num_nodes) {
    int i = blockIdx.x * blockDim.x + threadIdx.x;
    if (i < Bb && num_nodes[i] == -1.0f) num_nodes[i] = (float)MAXN;
}
// out = a + b, fp32 + bf16 copies
__global__ void add_kernel(const float* __restrict__ a, const float* __restrict__ b,
                           float* __restrict__ of, unsigned short* __restrict__ ob) {
    int i = blockIdx.x * blockDim.x + threadIdx.x;
    if (i < Bb * Hh) {
        float v = a[i] + b[i];
        of[i] = v;
        ob[i] = f2bf_bits(v);
    }
}

// ---------------- latent projection: h0 = selu(z @ lpW^T + b) ----------------
// grid (Hh/16, Bb/(16*8)), 256 threads (8 waves), 1 wave == one 16x16 D tile
__global__ void __launch_bounds__(256)
latproj_kernel(const unsigned short* __restrict__ zb,   // [B,128] bf16
               const unsigned short* __restrict__ Wt,   // [128,256] bf16 K-major
               const float* __restrict__ bias,
               float* __restrict__ hf, unsigned short* __restrict__ hb) {
    const int lane = threadIdx.x & 31;
    const int wave = threadIdx.x >> 5;
    const int mb = (blockIdx.y * 8 + wave) * 16;
    const int nb = blockIdx.x * 16;
    const int M = lane & 15, half = lane >> 4;
    const int rowA = mb + M;

    v8f acc = {};
    for (int kt = 0; kt < LATENT / 32; ++kt) {
        const int kb = kt * 32;
        v16bf a = load_frag_a(zb + rowA * LATENT + kb + half * 8);
        v16bf b = load_frag_b(Wt + (kb + half * 8) * Hh + nb + M, Hh);
        acc = wmma_bf16(a, b, acc);
    }
    const int col = nb + (lane & 15);
    const float bv = bias[col];
    const float kScale = 1.0507009873554805f, kAlpha = 1.6732632423543772f;
    for (int v = 0; v < 8; ++v) {
        int row = mb + v + 8 * half;
        float x = acc[v] + bv;
        float y = kScale * (x > 0.0f ? x : kAlpha * (__expf(x) - 1.0f));
        hf[row * Hh + col] = y;
        hb[row * Hh + col] = f2bf_bits(y);
    }
}

// ---------------- fused GRU step (both matmuls + gates), WMMA bf16 ----------------
// grid (Hh/16 = 16, 4), 256 threads = 8 waves; wave -> one 16x16 tile of h_new.
// The block's weight slab (both matrices, 3 gate column-groups, K=256) is
// block-uniform -> staged once into LDS (56KB) and fed to WMMA via strided 16-bit LDS reads,
// eliminating the 8x redundant per-wave global loads of round 1.
__global__ void __launch_bounds__(256)
gru_wmma_kernel(const unsigned short* __restrict__ xb,    // x   [B,H] bf16
                const unsigned short* __restrict__ hbin,  // h   [B,H] bf16
                const float*          __restrict__ hfin,  // h   [B,H] f32
                const unsigned short* __restrict__ Wih_t, // [H,3H] bf16 K-major
                const unsigned short* __restrict__ Whh_t, // [H,3H] bf16 K-major
                const float* __restrict__ bih, const float* __restrict__ bhh,
                float* __restrict__ hf_out, unsigned short* __restrict__ hb_out,
                float* __restrict__ extra_f,              // optional (edge_out slot)
                unsigned short* __restrict__ extra_b) {   // optional (node-emb ring)
    __shared__ unsigned short lds_w[2][Hh][WPAD];         // 2*256*56*2B = 56KB

    const int lane = threadIdx.x & 31;
    const int wave = threadIdx.x >> 5;
    const int mb = (blockIdx.y * 8 + wave) * 16;
    const int nb = blockIdx.x * 16;
    const int M = lane & 15, half = lane >> 4;
    const int rowA = mb + M;

    // --- cooperative stage: 1536 chunks of 32B (m in {ih,hh}, k in 0..255, g in 0..2)
    for (int ch = threadIdx.x; ch < 2 * 3 * Hh; ch += 256) {
        const int m   = ch / (3 * Hh);
        const int rem = ch - m * (3 * Hh);
        const int k   = rem & (Hh - 1);
        const int g   = rem >> 8;          // rem / 256
        const unsigned short* src = (m ? Whh_t : Wih_t) + (size_t)k * H3 + g * Hh + nb;
        const uint4* s = (const uint4*)src;
        uint4* d = (uint4*)(&lds_w[m][k][g * 16]);
        d[0] = s[0];
        d[1] = s[1];
    }
    __syncthreads();

    v8f a_ir = {}, a_iz = {}, a_in = {}, a_hr = {}, a_hz = {}, a_hn = {};

    for (int kt = 0; kt < Hh / 32; ++kt) {
        const int kb = kt * 32;
        v16bf ax = load_frag_a(xb   + rowA * Hh + kb + half * 8);
        v16bf ah = load_frag_a(hbin + rowA * Hh + kb + half * 8);
        const unsigned short* wi = &lds_w[0][kb + half * 8][M];
        const unsigned short* wh = &lds_w[1][kb + half * 8][M];
        v16bf b;
        b = load_frag_b(wi, WPAD);        a_ir = wmma_bf16(ax, b, a_ir);
        b = load_frag_b(wi + 16, WPAD);   a_iz = wmma_bf16(ax, b, a_iz);
        b = load_frag_b(wi + 32, WPAD);   a_in = wmma_bf16(ax, b, a_in);
        b = load_frag_b(wh, WPAD);        a_hr = wmma_bf16(ah, b, a_hr);
        b = load_frag_b(wh + 16, WPAD);   a_hz = wmma_bf16(ah, b, a_hz);
        b = load_frag_b(wh + 32, WPAD);   a_hn = wmma_bf16(ah, b, a_hn);
    }

    const int col = nb + (lane & 15);           // C/D: N = lane&15, M = v + 8*half
    const float b_ir = bih[col], b_iz = bih[Hh + col], b_in = bih[2 * Hh + col];
    const float b_hr = bhh[col], b_hz = bhh[Hh + col], b_hn = bhh[2 * Hh + col];
    for (int v = 0; v < 8; ++v) {
        const int row = mb + v + 8 * half;
        float r  = sigmoidf((a_ir[v] + b_ir) + (a_hr[v] + b_hr));
        float zg = sigmoidf((a_iz[v] + b_iz) + (a_hz[v] + b_hz));
        float nn = tanhf((a_in[v] + b_in) + r * (a_hn[v] + b_hn));
        float ho = hfin[row * Hh + col];
        float hn = (1.0f - zg) * nn + zg * ho;
        unsigned short hb16 = f2bf_bits(hn);
        hf_out[row * Hh + col] = hn;
        hb_out[row * Hh + col] = hb16;
        if (extra_f) extra_f[row * Hh + col] = hn;
        if (extra_b) extra_b[row * Hh + col] = hb16;
    }
}

// ---------------- node classifiers + gumbel + stop logic ----------------
// one block (64 thr) per batch row; 51 dot products of length 256
__global__ void node_cls_kernel(const float* __restrict__ emb,
                                const float* __restrict__ W0, const float* __restrict__ c0,
                                const float* __restrict__ W1, const float* __restrict__ c1,
                                const float* __restrict__ W2, const float* __restrict__ c2,
                                float* __restrict__ out_feats,   // [B,50,51]
                                float* __restrict__ out_lp,      // [B,50,3]
                                float* __restrict__ num_nodes, int step) {
    __shared__ float se[Hh];
    __shared__ float slog[51];
    const int b = blockIdx.x, t = threadIdx.x;
    for (int k = t; k < Hh; k += 64) se[k] = emb[b * Hh + k];
    __syncthreads();
    const int widths[3] = {40, 6, 5};
    const int cum[3]    = {0, 40, 46};
    if (t < 51) {
        int j  = (t < 40) ? 0 : (t < 46 ? 1 : 2);
        int cl = t - cum[j];
        const float* w = (j == 0 ? W0 : (j == 1 ? W1 : W2)) + cl * Hh;
        float acc = (j == 0 ? c0 : (j == 1 ? c1 : c2))[cl];
        for (int k = 0; k < Hh; ++k) acc += se[k] * w[k];
        slog[t] = acc;
    }
    __syncthreads();
    if (t < 3) {
        const int j = t, nf = widths[j], base = cum[j];
        float mx = -1e30f;
        for (int c = 0; c < nf; ++c) mx = fmaxf(mx, slog[base + c]);
        float sum = 0.0f;
        for (int c = 0; c < nf; ++c) sum += __expf(slog[base + c] - mx);
        const float lse = mx + __logf(sum);
        // gumbel-softmax sample (temp 1e-3)
        const unsigned long long seed = (unsigned long long)(step * 100 + j);
        float y[40];
        float ymx = -1e30f; int amax = 0;
        for (int c = 0; c < nf; ++c) {
            float g  = gumbel(seed, (unsigned long long)(b * 64 + c));
            float yy = (slog[base + c] + g) * INV_TEMP;
            y[c] = yy;
            if (yy > ymx) { ymx = yy; amax = c; }
        }
        float ss = 0.0f;
        for (int c = 0; c < nf; ++c) { y[c] = __expf(y[c] - ymx); ss += y[c]; }
        const float inv = 1.0f / ss;
        float lp = 0.0f;
        float* of = out_feats + ((size_t)b * MAXN + step) * 51 + base;
        for (int c = 0; c < nf; ++c) {
            float s = y[c] * inv;
            of[c] = s;
            lp += s * (slog[base + c] - lse);       // sum(sg(s) * log_softmax(logits))
        }
        out_lp[((size_t)b * MAXN + step) * 3 + j] = lp;
        if (j == 0 && amax == END_NODE && num_nodes[b] == -1.0f)
            num_nodes[b] = (float)(step + 1);
    }
}

// ---------------- edge classifiers (logp of the SAMPLE, per reference) ----------
// grid (B, L); blockIdx.y = scan slot t, reversed row block r = L-1-t
__global__ void edge_cls_kernel(const float* __restrict__ eout, int slot_stride,
                                const float* __restrict__ W0, const float* __restrict__ c0,
                                const float* __restrict__ W1, const float* __restrict__ c1,
                                float* __restrict__ out_feats,   // [522,B,9]
                                float* __restrict__ out_lp,      // [522,B,2]
                                int base_row, int L, int step) {
    __shared__ float se[Hh];
    __shared__ float slog[9];
    const int b = blockIdx.x, tscan = blockIdx.y, t = threadIdx.x;
    const int r = L - 1 - tscan;
    const float* src = eout + (size_t)tscan * slot_stride + (size_t)b * Hh;
    for (int k = t; k < Hh; k += 64) se[k] = src[k];
    __syncthreads();
    const int widths[2] = {5, 4};
    const int cum[2]    = {0, 5};
    if (t < 9) {
        int j = (t < 5) ? 0 : 1;
        int cl = t - cum[j];
        const float* w = (j ? W1 : W0) + cl * Hh;
        float acc = (j ? c1 : c0)[cl];
        for (int k = 0; k < Hh; ++k) acc += se[k] * w[k];
        slog[t] = acc;
    }
    __syncthreads();
    if (t < 2) {
        const int j = t, nf = widths[j], base = cum[j];
        const int flat_row = r * Bb + b;
        const unsigned long long seed = (unsigned long long)(step * 100 + 50 + j);
        float y[5];
        float ymx = -1e30f;
        for (int c = 0; c < nf; ++c) {
            float g  = gumbel(seed, (unsigned long long)(flat_row * 16 + c));
            float yy = (slog[base + c] + g) * INV_TEMP;
            y[c] = yy;
            ymx = fmaxf(ymx, yy);
        }
        float ss = 0.0f;
        for (int c = 0; c < nf; ++c) { y[c] = __expf(y[c] - ymx); ss += y[c]; }
        const float inv = 1.0f / ss;
        // s = sample; logp = log_softmax(s) (faithful to source)
        float smx = -1e30f;
        for (int c = 0; c < nf; ++c) { y[c] *= inv; smx = fmaxf(smx, y[c]); }
        float s2 = 0.0f;
        for (int c = 0; c < nf; ++c) s2 += __expf(y[c] - smx);
        const float lse2 = smx + __logf(s2);
        const int grow = base_row + r;
        float* of = out_feats + ((size_t)grow * Bb + b) * 9 + base;
        float lp = 0.0f;
        for (int c = 0; c < nf; ++c) {
            of[c] = y[c];
            lp += y[c] * (y[c] - lse2);
        }
        out_lp[((size_t)grow * Bb + b) * 2 + j] = lp;
    }
}

// ---------------- host orchestration ----------------
extern "C" void kernel_launch(void* const* d_in, const int* in_sizes, int n_in,
                              void* d_out, int out_size, void* d_ws, size_t ws_size,
                              hipStream_t stream) {
    (void)in_sizes; (void)n_in; (void)out_size;
    const float* z_in  = (const float*)d_in[0];
    const float* lp_W  = (const float*)d_in[1];
    const float* lp_b  = (const float*)d_in[2];
    const float* n_Wih = (const float*)d_in[3];
    const float* n_Whh = (const float*)d_in[4];
    const float* n_bih = (const float*)d_in[5];
    const float* n_bhh = (const float*)d_in[6];
    const float* e_Wih = (const float*)d_in[7];
    const float* e_Whh = (const float*)d_in[8];
    const float* e_bih = (const float*)d_in[9];
    const float* e_bhh = (const float*)d_in[10];
    const float* ncW0 = (const float*)d_in[11]; const float* ncb0 = (const float*)d_in[12];
    const float* ncW1 = (const float*)d_in[13]; const float* ncb1 = (const float*)d_in[14];
    const float* ncW2 = (const float*)d_in[15]; const float* ncb2 = (const float*)d_in[16];
    const float* ecW0 = (const float*)d_in[17]; const float* ecb0 = (const float*)d_in[18];
    const float* ecW1 = (const float*)d_in[19]; const float* ecb1 = (const float*)d_in[20];

    // --- workspace carve-up (256B-aligned slabs) ---
    char* ws = (char*)d_ws;
    size_t off = 0;
    auto carve = [&](size_t bytes) -> char* {
        char* p = ws + off;
        off += (bytes + 255) & ~(size_t)255;
        return p;
    };
    unsigned short* Wt_lp  = (unsigned short*)carve((size_t)LATENT * Hh * 2);
    unsigned short* Wt_nih = (unsigned short*)carve((size_t)Hh * H3 * 2);
    unsigned short* Wt_nhh = (unsigned short*)carve((size_t)Hh * H3 * 2);
    unsigned short* Wt_eih = (unsigned short*)carve((size_t)Hh * H3 * 2);
    unsigned short* Wt_ehh = (unsigned short*)carve((size_t)Hh * H3 * 2);
    unsigned short* z_bf   = (unsigned short*)carve((size_t)Bb * LATENT * 2);
    unsigned short* x0     = (unsigned short*)carve((size_t)Bb * Hh * 2);
    unsigned short* nh_bf[2], *eh_bf[2];
    nh_bf[0] = (unsigned short*)carve((size_t)Bb * Hh * 2);
    nh_bf[1] = (unsigned short*)carve((size_t)Bb * Hh * 2);
    eh_bf[0] = (unsigned short*)carve((size_t)Bb * Hh * 2);
    eh_bf[1] = (unsigned short*)carve((size_t)Bb * Hh * 2);
    unsigned short* ring[RING];
    for (int k = 0; k < RING; ++k) ring[k] = (unsigned short*)carve((size_t)Bb * Hh * 2);
    float* nh_f[2]; float* eh_f[2];
    nh_f[0] = (float*)carve((size_t)Bb * Hh * 4);
    nh_f[1] = (float*)carve((size_t)Bb * Hh * 4);
    eh_f[0] = (float*)carve((size_t)Bb * Hh * 4);
    eh_f[1] = (float*)carve((size_t)Bb * Hh * 4);
    float* eout = (float*)carve((size_t)EDGE_W * Bb * Hh * 4);
    if (off > ws_size) return;   // workspace too small: safe no-op

    // --- output layout (tuple flattened in return order) ---
    float* out   = (float*)d_out;
    float* o_num = out;                               // [512]
    float* o_nf  = o_num + Bb;                        // [B,50,51]
    float* o_ef  = o_nf + (size_t)Bb * MAXN * 51;     // [522,B,9]
    float* o_nlp = o_ef + (size_t)522 * Bb * 9;       // [B,50,3]
    float* o_elp = o_nlp + (size_t)Bb * MAXN * 3;     // [522,B,2]

    // --- weight conversion (bf16, K-major transpose) ---
    {
        int n;
        n = Hh * LATENT; convT_kernel<<<(n + 255) / 256, 256, 0, stream>>>(lp_W, Wt_lp, Hh, LATENT);
        n = H3 * Hh;
        convT_kernel<<<(n + 255) / 256, 256, 0, stream>>>(n_Wih, Wt_nih, H3, Hh);
        convT_kernel<<<(n + 255) / 256, 256, 0, stream>>>(n_Whh, Wt_nhh, H3, Hh);
        convT_kernel<<<(n + 255) / 256, 256, 0, stream>>>(e_Wih, Wt_eih, H3, Hh);
        convT_kernel<<<(n + 255) / 256, 256, 0, stream>>>(e_Whh, Wt_ehh, H3, Hh);
        n = Bb * LATENT; conv_kernel<<<(n + 255) / 256, 256, 0, stream>>>(z_in, z_bf, n);
    }
    init_kernel<<<(Bb * Hh) / 256, 256, 0, stream>>>(o_num, x0, eh_f[0], eh_bf[0]);

    const dim3 mmGrid(Hh / 16, Bb / (16 * 8));   // (16, 4), 8 waves/block
    latproj_kernel<<<mmGrid, 256, 0, stream>>>(z_bf, Wt_lp, lp_b, nh_f[0], nh_bf[0]);

    int ncur = 0, ecur = 0;
    int edge_base = 0;
    for (int i = 0; i < MAXN; ++i) {
        const unsigned short* xin = (i == 0) ? x0 : ring[(i - 1) % RING];
        gru_wmma_kernel<<<mmGrid, 256, 0, stream>>>(
            xin, nh_bf[ncur], nh_f[ncur], Wt_nih, Wt_nhh, n_bih, n_bhh,
            nh_f[1 - ncur], nh_bf[1 - ncur], nullptr, ring[i % RING]);
        ncur = 1 - ncur;

        node_cls_kernel<<<Bb, 64, 0, stream>>>(nh_f[ncur],
            ncW0, ncb0, ncW1, ncb1, ncW2, ncb2, o_nf, o_nlp, o_num, i);

        if (i > 0) {
            const int L = (i < EDGE_W) ? i : EDGE_W;
            // edge_h += node_h (pre-scan)
            add_kernel<<<(Bb * Hh) / 256, 256, 0, stream>>>(
                eh_f[ecur], nh_f[ncur], eh_f[1 - ecur], eh_bf[1 - ecur]);
            ecur = 1 - ecur;
            // sequential edge GRU scan over the last L node embeddings
            for (int t = 0; t < L; ++t) {
                gru_wmma_kernel<<<mmGrid, 256, 0, stream>>>(
                    ring[(i - 1 - t) % RING], eh_bf[ecur], eh_f[ecur],
                    Wt_eih, Wt_ehh, e_bih, e_bhh,
                    eh_f[1 - ecur], eh_bf[1 - ecur],
                    eout + (size_t)t * Bb * Hh, nullptr);
                ecur = 1 - ecur;
            }
            edge_cls_kernel<<<dim3(Bb, L), 64, 0, stream>>>(
                eout, Bb * Hh, ecW0, ecb0, ecW1, ecb1,
                o_ef, o_elp, edge_base, L, i);
            // node_h = node_state + edge_h (post-scan)
            add_kernel<<<(Bb * Hh) / 256, 256, 0, stream>>>(
                nh_f[ncur], eh_f[ecur], nh_f[1 - ncur], nh_bf[1 - ncur]);
            ncur = 1 - ncur;
            edge_base += L;
        }
    }
    finalize_kernel<<<(Bb + 255) / 256, 256, 0, stream>>>(o_num);
}
